// MambaBlock2D_25451976196545
// MI455X (gfx1250) — matrix operand order repaired and checked by reading, not compiled, not run on hardware
//
#include <hip/hip_runtime.h>
#include <hip/hip_bf16.h>
#include <math.h>

// ---------------------------------------------------------------------------
// Types for CDNA5 WMMA (wave32): 16x16x32 bf16 -> f32
// ---------------------------------------------------------------------------
typedef __bf16 bf16x16 __attribute__((ext_vector_type(16)));
typedef float  f32x8   __attribute__((ext_vector_type(8)));
typedef unsigned int u32x4 __attribute__((ext_vector_type(4)));
typedef unsigned int u32x8 __attribute__((ext_vector_type(8)));

union FragAB {
    bf16x16      v;
    unsigned int u[8];
};

static constexpr int D_MODEL = 256;
static constexpr int D_INNER = 512;
static constexpr int D_STATE = 16;
static constexpr int SEQ_L   = 64;
static constexpr int NSEQ    = 8 * 64;          // b * h (or b * w)  = 512
static constexpr int NTOK    = NSEQ * SEQ_L;    // 32768 tokens per direction

// ---------------------------------------------------------------------------
// Helpers
// ---------------------------------------------------------------------------
__device__ __forceinline__ unsigned short f2bf(float f) {
    unsigned int u = __builtin_bit_cast(unsigned int, f);
    u += 0x7fffu + ((u >> 16) & 1u);   // round-to-nearest-even
    return (unsigned short)(u >> 16);
}
__device__ __forceinline__ float bf2f(unsigned short s) {
    unsigned int u = ((unsigned int)s) << 16;
    return __builtin_bit_cast(float, u);
}
__device__ __forceinline__ float silu_f(float v) {
    return v / (1.0f + __expf(-v));
}
__device__ __forceinline__ float softplus_f(float v) {
    return v > 20.0f ? v : log1pf(__expf(v));
}
// Low 32 bits of a flat pointer into LDS == LDS byte address (shared aperture
// occupies addr[63:32] only; addr[31:0] is the LDS offset).
__device__ __forceinline__ unsigned lds_addr_of(const void* p) {
    return (unsigned)(unsigned long long)p;
}
__device__ __forceinline__ unsigned rfl(unsigned x) {
    return (unsigned)__builtin_amdgcn_readfirstlane((int)x);
}

// A fragment: 16(M) x 32(K) bf16, row-major source, row stride = K elements.
// ISA layout: lanes 0-15 -> M=lane, K={0..7,16..23}; lanes 16-31 -> M=lane-16,
// K={8..15,24..31}; each VGPR holds 2 consecutive K (packed 16-bit).
__device__ __forceinline__ void load_A_frag(FragAB& a, const unsigned short* A,
                                            int row0, int K, int kb, int lane) {
    const int r    = lane & 15;
    const int koff = (lane < 16) ? 0 : 8;
    const unsigned short* p = A + (size_t)(row0 + r) * K + kb;
#pragma unroll
    for (int j = 0; j < 4; ++j) {
        a.u[j]     = *reinterpret_cast<const unsigned int*>(p + koff + 2 * j);
        a.u[4 + j] = *reinterpret_cast<const unsigned int*>(p + 16 + koff + 2 * j);
    }
}

// B fragment from an LDS panel laid out N-major: Bs[n_local*K + k].
// ISA layout: lanes 0-15 -> N=lane, K={0..7,16..23}; lanes 16-31 -> N=lane-16, K={8..15,24..31}.
__device__ __forceinline__ void load_B_frag_lds(FragAB& b, const unsigned short* Bs,
                                                int nl0, int K, int kb, int lane) {
    const int n    = nl0 + (lane & 15);
    const int koff = (lane < 16) ? 0 : 8;
    const unsigned short* p = Bs + n * K + kb + koff;
#pragma unroll
    for (int j = 0; j < 4; ++j) {
        b.u[j]     = *reinterpret_cast<const unsigned int*>(p + 2 * j);
        b.u[4 + j] = *reinterpret_cast<const unsigned int*>(p + 16 + 2 * j);
    }
}

// ---------------------------------------------------------------------------
// Async staging: memory -> LDS with no VGPR round trip.
// Each lane issues global_load_async_to_lds_b128 (ASYNCcnt-tracked), then the
// wave drains its ASYNCcnt and the block barriers.
// ---------------------------------------------------------------------------
template <int NCOLS, int K>
__device__ __forceinline__ void stage_B_async(unsigned short* Bs,
                                              const unsigned short* __restrict__ Wt,
                                              int n0) {
    constexpr int total_bytes = NCOLS * K * 2;
    const char* src = (const char*)(Wt + (size_t)n0 * K);
    const unsigned lds_base = lds_addr_of(Bs);
#pragma unroll 1
    for (int byte = threadIdx.x * 16; byte < total_bytes; byte += 256 * 16) {
        unsigned lds_a = lds_base + byte;
        unsigned long long ga = (unsigned long long)(src + byte);
        asm volatile("global_load_async_to_lds_b128 %0, %1, off"
                     :: "v"(lds_a), "v"(ga) : "memory");
    }
    asm volatile("s_wait_asynccnt 0x0" ::: "memory");
    __syncthreads();
}

// ---------------------------------------------------------------------------
// TDM staging: one wave posts a 1-D tile D# (groups 0+1 per ISA §8.3/8.4),
// the Tensor Data Mover DMAs the whole contiguous panel into LDS, and we wait
// on TENSORcnt.  nelems = number of bf16 elements (fits tile_dim0 <= 65535).
// ---------------------------------------------------------------------------
__device__ __forceinline__ void tdm_stage_panel(unsigned short* Bs,
                                                const unsigned short* __restrict__ gsrc,
                                                unsigned nelems) {
    if ((threadIdx.x >> 5) == 0) {            // wave 0 issues; EXEC ignored by TDM
        unsigned long long ga = (unsigned long long)gsrc;
        unsigned lds_base = lds_addr_of(Bs);
        u32x4 g0;
        g0[0] = rfl(1u);                                // count=1 (valid user D#)
        g0[1] = rfl(lds_base);                          // lds_addr [63:32]
        g0[2] = rfl((unsigned)ga);                      // global_addr[31:0]
        g0[3] = rfl((unsigned)(ga >> 32) | (2u << 30)); // global_addr[56:32] | type=2
        u32x8 g1;
        g1[0] = rfl(1u << 16);                          // data_size=1 (2 bytes)
        g1[1] = rfl((nelems & 0xffffu) << 16);          // tensor_dim0[15:0] @ bits 63:48
        g1[2] = rfl((nelems >> 16) | (1u << 16));       // tensor_dim0[31:16] | tensor_dim1=1
        g1[3] = rfl((nelems & 0xffffu) << 16);          // tile_dim0 @ bits 127:112
        g1[4] = rfl(0u);                                // tile_dim1/2 unused
        g1[5] = rfl(nelems);                            // tensor_dim0_stride[31:0]
        g1[6] = rfl(0u);
        g1[7] = rfl(0u);
        asm volatile("tensor_load_to_lds %0, %1" :: "s"(g0), "s"(g1) : "memory");
        __builtin_amdgcn_s_wait_tensorcnt(0);
    }
    __syncthreads();
}

#define WMMA_BF16(acc, a, b) \
    __builtin_amdgcn_wmma_f32_16x16x32_bf16(false, (a).v, false, (b).v, (short)0, (acc), false, false)

// ---------------------------------------------------------------------------
// 0) Weight transpose + f32->bf16 (with optional K zero-padding)
//    src: W[K x N] row-major f32.  dst: Wt[n*Kp + k] bf16, zeros for k >= K.
// ---------------------------------------------------------------------------
__global__ void convert_wT_kernel(const float* __restrict__ W,
                                  unsigned short* __restrict__ Wt,
                                  int K, int N, int Kp) {
    int idx = blockIdx.x * blockDim.x + threadIdx.x;
    if (idx >= N * Kp) return;
    int k = idx % Kp;
    int n = idx / Kp;
    Wt[idx] = (k < K) ? f2bf(W[(size_t)k * N + n]) : (unsigned short)0;
}

// ---------------------------------------------------------------------------
// 1) Gather x (B, C, H, W) into per-direction token matrix U: [NTOK x 256] bf16
//    dir 0: seq = b*64 + h, pos = w.   dir 1: seq = b*64 + w, pos = h.
// ---------------------------------------------------------------------------
__global__ void pack_tokens_kernel(const float* __restrict__ x,
                                   unsigned short* __restrict__ U, int dir) {
    size_t idx = (size_t)blockIdx.x * blockDim.x + threadIdx.x;  // t*256 + c
    int c   = (int)(idx & 255);
    size_t t = idx >> 8;
    int pos = (int)(t & 63);
    int seq = (int)(t >> 6);
    int b   = seq >> 6;
    int q   = seq & 63;
    int h   = (dir == 0) ? q : pos;
    int w   = (dir == 0) ? pos : q;
    float v = x[((((size_t)b * 256 + c) * 64 + h) * 64) + w];
    U[idx] = f2bf(v);
}

// ---------------------------------------------------------------------------
// 2) xz = U @ W_in  (M=NTOK, K=256, N=1024), split into X (n<512) / Z buffers.
//    Weight panel (64 x 256 bf16 = 32 KB) async-staged once in LDS, shared by
//    all 8 waves.
// ---------------------------------------------------------------------------
__global__ void gemm_xz_kernel(const unsigned short* __restrict__ U,
                               const unsigned short* __restrict__ WinT,
                               float* __restrict__ X, float* __restrict__ Z) {
    constexpr int K = 256;
    __shared__ __align__(16) unsigned short Bs[64 * K];

    const int wave = threadIdx.x >> 5;
    const int lane = threadIdx.x & 31;
    const int m0   = blockIdx.x * 128 + wave * 16;
    const int n0   = blockIdx.y * 64;

    stage_B_async<64, K>(Bs, WinT, n0);

    f32x8 acc0 = {}, acc1 = {}, acc2 = {}, acc3 = {};
    for (int kb = 0; kb < K; kb += 32) {
        if (kb + 32 < K)
            __builtin_prefetch(U + (size_t)(m0 + (lane & 15)) * K + kb + 32, 0, 1);
        FragAB a; load_A_frag(a, U, m0, K, kb, lane);
        FragAB b;
        load_B_frag_lds(b, Bs,  0, K, kb, lane); acc0 = WMMA_BF16(acc0, a, b);
        load_B_frag_lds(b, Bs, 16, K, kb, lane); acc1 = WMMA_BF16(acc1, a, b);
        load_B_frag_lds(b, Bs, 32, K, kb, lane); acc2 = WMMA_BF16(acc2, a, b);
        load_B_frag_lds(b, Bs, 48, K, kb, lane); acc3 = WMMA_BF16(acc3, a, b);
    }
    const int c      = lane & 15;
    const int rowoff = (lane < 16) ? 0 : 8;
    f32x8 accs[4] = {acc0, acc1, acc2, acc3};
#pragma unroll
    for (int t = 0; t < 4; ++t) {
        int n = n0 + 16 * t + c;
        float* dst = (n < D_INNER) ? X : Z;
        int nn = n & (D_INNER - 1);
#pragma unroll
        for (int r = 0; r < 8; ++r) {
            int m = m0 + rowoff + r;
            dst[(size_t)m * D_INNER + nn] = accs[t][r];
        }
    }
}

// ---------------------------------------------------------------------------
// 3) Depthwise causal conv (width 4) + bias + SiLU -> XC (bf16)
// ---------------------------------------------------------------------------
__global__ void conv_silu_kernel(const float* __restrict__ X,
                                 const float* __restrict__ conv_w,
                                 const float* __restrict__ conv_b,
                                 unsigned short* __restrict__ XCb) {
    size_t idx = (size_t)blockIdx.x * blockDim.x + threadIdx.x;  // t*512 + d
    int d    = (int)(idx & (D_INNER - 1));
    size_t t = idx >> 9;
    int pos  = (int)(t & (SEQ_L - 1));
    float acc = conv_b[d];
#pragma unroll
    for (int j = 0; j < 4; ++j) {
        int back = 3 - j;
        if (pos - back >= 0)
            acc += conv_w[d * 4 + j] * X[(t - back) * D_INNER + d];
    }
    XCb[idx] = f2bf(silu_f(acc));
}

// ---------------------------------------------------------------------------
// 4) dbc = XC @ W_xproj  (K=512, N=48 -> three 16-wide tiles)
//    tile 0 -> DTIN bf16 [M x 32] zero-padded, tile 1 -> Bm, tile 2 -> Cm.
//    16 x 512 bf16 = 16 KB weight panel async-staged in LDS.
// ---------------------------------------------------------------------------
__global__ void gemm_xproj_kernel(const unsigned short* __restrict__ XCb,
                                  const unsigned short* __restrict__ WxT,
                                  unsigned short* __restrict__ DTIN,
                                  float* __restrict__ Bm, float* __restrict__ Cm) {
    constexpr int K = D_INNER;
    __shared__ __align__(16) unsigned short Bs[16 * K];

    const int wave = threadIdx.x >> 5;
    const int lane = threadIdx.x & 31;
    const int m0   = blockIdx.x * 128 + wave * 16;
    const int nt   = blockIdx.y;

    stage_B_async<16, K>(Bs, WxT, nt * 16);

    f32x8 acc = {};
    for (int kb = 0; kb < K; kb += 32) {
        if (kb + 32 < K)
            __builtin_prefetch(XCb + (size_t)(m0 + (lane & 15)) * K + kb + 32, 0, 1);
        FragAB a; load_A_frag(a, XCb, m0, K, kb, lane);
        FragAB b; load_B_frag_lds(b, Bs, 0, K, kb, lane);
        acc = WMMA_BF16(acc, a, b);
    }
    const int c      = lane & 15;
    const int rowoff = (lane < 16) ? 0 : 8;
#pragma unroll
    for (int r = 0; r < 8; ++r) {
        int m = m0 + rowoff + r;
        float v = acc[r];
        if (nt == 0) {
            DTIN[(size_t)m * 32 + c]      = f2bf(v);
            DTIN[(size_t)m * 32 + 16 + c] = 0;          // K-pad
        } else if (nt == 1) {
            Bm[(size_t)m * D_STATE + c] = v;
        } else {
            Cm[(size_t)m * D_STATE + c] = v;
        }
    }
}

// ---------------------------------------------------------------------------
// 5) dt = softplus(DTIN @ W_dt + b_dt)   (K=32 incl. zero pad, N=512)
// ---------------------------------------------------------------------------
__global__ void gemm_dt_kernel(const unsigned short* __restrict__ DTIN,
                               const unsigned short* __restrict__ WdtT,
                               const float* __restrict__ b_dt,
                               float* __restrict__ DT) {
    constexpr int K = 32;
    __shared__ __align__(16) unsigned short Bs[64 * K];

    const int wave = threadIdx.x >> 5;
    const int lane = threadIdx.x & 31;
    const int m0   = blockIdx.x * 128 + wave * 16;
    const int n0   = blockIdx.y * 64;

    stage_B_async<64, K>(Bs, WdtT, n0);

    FragAB a; load_A_frag(a, DTIN, m0, K, 0, lane);
    f32x8 acc0 = {}, acc1 = {}, acc2 = {}, acc3 = {};
    FragAB b;
    load_B_frag_lds(b, Bs,  0, K, 0, lane); acc0 = WMMA_BF16(acc0, a, b);
    load_B_frag_lds(b, Bs, 16, K, 0, lane); acc1 = WMMA_BF16(acc1, a, b);
    load_B_frag_lds(b, Bs, 32, K, 0, lane); acc2 = WMMA_BF16(acc2, a, b);
    load_B_frag_lds(b, Bs, 48, K, 0, lane); acc3 = WMMA_BF16(acc3, a, b);

    const int c      = lane & 15;
    const int rowoff = (lane < 16) ? 0 : 8;
    f32x8 accs[4] = {acc0, acc1, acc2, acc3};
#pragma unroll
    for (int t = 0; t < 4; ++t) {
        int n = n0 + 16 * t + c;
        float bias = b_dt[n];
#pragma unroll
        for (int r = 0; r < 8; ++r) {
            int m = m0 + rowoff + r;
            DT[(size_t)m * D_INNER + n] = softplus_f(accs[t][r] + bias);
        }
    }
}

// ---------------------------------------------------------------------------
// 6) Selective scan, fused with skip (x*D) and silu(z) gating -> Y bf16
// ---------------------------------------------------------------------------
__global__ void scan_kernel(const float* __restrict__ DT,
                            const float* __restrict__ Bm,
                            const float* __restrict__ Cm,
                            const unsigned short* __restrict__ XCb,
                            const float* __restrict__ Z,
                            const float* __restrict__ A_log,
                            const float* __restrict__ Dp,
                            unsigned short* __restrict__ Yb) {
    int idx = blockIdx.x * blockDim.x + threadIdx.x;   // seq*512 + d
    int d   = idx & (D_INNER - 1);
    int seq = idx >> 9;

    float A[D_STATE];
#pragma unroll
    for (int n = 0; n < D_STATE; ++n)
        A[n] = -__expf(A_log[(size_t)d * D_STATE + n]);

    float h[D_STATE];
#pragma unroll
    for (int n = 0; n < D_STATE; ++n) h[n] = 0.0f;

    const float dskip = Dp[d];
    const size_t tbase = (size_t)seq * SEQ_L;

    for (int pos = 0; pos < SEQ_L; ++pos) {
        size_t t   = tbase + pos;
        float dt   = DT[t * D_INNER + d];
        float xv   = bf2f(XCb[t * D_INNER + d]);
        float dtx  = dt * xv;
        const float* Bt = Bm + t * D_STATE;
        const float* Ct = Cm + t * D_STATE;
        float y = 0.0f;
#pragma unroll
        for (int n = 0; n < D_STATE; ++n) {
            float e = __expf(dt * A[n]);
            h[n] = e * h[n] + dtx * Bt[n];
            y   += h[n] * Ct[n];
        }
        float z   = Z[t * D_INNER + d];
        float out = (y + xv * dskip) * silu_f(z);
        Yb[t * D_INNER + d] = f2bf(out);
    }
}

// ---------------------------------------------------------------------------
// 7) out = Y @ W_out  (K=512, N=256), scattered back to (B,C,H,W).
//    64 x 512 bf16 = 64 KB weight panel staged by the Tensor Data Mover.
//    dir 0 stores (covers every element), dir 1 accumulates.
// ---------------------------------------------------------------------------
__global__ void gemm_out_kernel(const unsigned short* __restrict__ Yb,
                                const unsigned short* __restrict__ WoutT,
                                float* __restrict__ out, int dir) {
    constexpr int K = D_INNER;
    __shared__ __align__(16) unsigned short Bs[64 * K];

    const int wave = threadIdx.x >> 5;
    const int lane = threadIdx.x & 31;
    const int m0   = blockIdx.x * 128 + wave * 16;
    const int n0   = blockIdx.y * 64;

    tdm_stage_panel(Bs, WoutT + (size_t)n0 * K, 64 * K);

    f32x8 acc0 = {}, acc1 = {}, acc2 = {}, acc3 = {};
    for (int kb = 0; kb < K; kb += 32) {
        if (kb + 32 < K)
            __builtin_prefetch(Yb + (size_t)(m0 + (lane & 15)) * K + kb + 32, 0, 1);
        FragAB a; load_A_frag(a, Yb, m0, K, kb, lane);
        FragAB b;
        load_B_frag_lds(b, Bs,  0, K, kb, lane); acc0 = WMMA_BF16(acc0, a, b);
        load_B_frag_lds(b, Bs, 16, K, kb, lane); acc1 = WMMA_BF16(acc1, a, b);
        load_B_frag_lds(b, Bs, 32, K, kb, lane); acc2 = WMMA_BF16(acc2, a, b);
        load_B_frag_lds(b, Bs, 48, K, kb, lane); acc3 = WMMA_BF16(acc3, a, b);
    }
    const int c      = lane & 15;
    const int rowoff = (lane < 16) ? 0 : 8;
    f32x8 accs[4] = {acc0, acc1, acc2, acc3};
#pragma unroll
    for (int t = 0; t < 4; ++t) {
        int n = n0 + 16 * t + c;   // channel
#pragma unroll
        for (int r = 0; r < 8; ++r) {
            int m   = m0 + rowoff + r;  // token
            int pos = m & 63;
            int seq = m >> 6;
            int b   = seq >> 6;
            int q   = seq & 63;
            int hh  = (dir == 0) ? q : pos;
            int ww  = (dir == 0) ? pos : q;
            size_t o = (((size_t)b * 256 + n) * 64 + hh) * 64 + ww;
            if (dir == 0) out[o] = accs[t][r];
            else          out[o] += accs[t][r];
        }
    }
}

// ---------------------------------------------------------------------------
// Launch
// ---------------------------------------------------------------------------
extern "C" void kernel_launch(void* const* d_in, const int* in_sizes, int n_in,
                              void* d_out, int out_size, void* d_ws, size_t ws_size,
                              hipStream_t stream) {
    (void)in_sizes; (void)n_in; (void)out_size; (void)ws_size;

    const float* x = (const float*)d_in[0];
    float* out = (float*)d_out;

    // ---- workspace layout (bytes) ----
    char* ws = (char*)d_ws;
    size_t off = 0;
    auto alloc = [&](size_t bytes) { void* p = ws + off; off += (bytes + 255) & ~size_t(255); return p; };

    unsigned short* U     = (unsigned short*)alloc((size_t)NTOK * D_MODEL * 2);   // 16.8 MB
    float*          XDT   = (float*)alloc((size_t)NTOK * D_INNER * 4);            // 67 MB (X, later DT)
    float*          Zbuf  = (float*)alloc((size_t)NTOK * D_INNER * 4);            // 67 MB
    unsigned short* XCb   = (unsigned short*)alloc((size_t)NTOK * D_INNER * 2);   // 33.5 MB
    unsigned short* DTIN  = (unsigned short*)alloc((size_t)NTOK * 32 * 2);        // 2.1 MB
    float*          Bm    = (float*)alloc((size_t)NTOK * D_STATE * 4);            // 2.1 MB
    float*          Cm    = (float*)alloc((size_t)NTOK * D_STATE * 4);            // 2.1 MB
    unsigned short* Yb    = (unsigned short*)alloc((size_t)NTOK * D_INNER * 2);   // 33.5 MB
    unsigned short* WinT  = (unsigned short*)alloc((size_t)1024 * 256 * 2);
    unsigned short* WxT   = (unsigned short*)alloc((size_t)48 * 512 * 2);
    unsigned short* WdtT  = (unsigned short*)alloc((size_t)512 * 32 * 2);
    unsigned short* WoutT = (unsigned short*)alloc((size_t)256 * 512 * 2);

    for (int dir = 0; dir < 2; ++dir) {
        const int base = 1 + dir * 9;
        const float* W_in    = (const float*)d_in[base + 0];   // 256 x 1024
        const float* conv_w  = (const float*)d_in[base + 1];   // 512 x 4
        const float* conv_b  = (const float*)d_in[base + 2];   // 512
        const float* W_xproj = (const float*)d_in[base + 3];   // 512 x 48
        const float* W_dt    = (const float*)d_in[base + 4];   // 16 x 512
        const float* b_dt    = (const float*)d_in[base + 5];   // 512
        const float* A_log   = (const float*)d_in[base + 6];   // 512 x 16
        const float* Dp      = (const float*)d_in[base + 7];   // 512
        const float* W_out   = (const float*)d_in[base + 8];   // 512 x 256

        // weights -> bf16, N-major (K-padded for W_dt)
        convert_wT_kernel<<<(1024 * 256 + 255) / 256, 256, 0, stream>>>(W_in,    WinT,  256, 1024, 256);
        convert_wT_kernel<<<(48 * 512 + 255) / 256,   256, 0, stream>>>(W_xproj, WxT,   512, 48,   512);
        convert_wT_kernel<<<(512 * 32 + 255) / 256,   256, 0, stream>>>(W_dt,    WdtT,  16,  512,  32);
        convert_wT_kernel<<<(256 * 512 + 255) / 256,  256, 0, stream>>>(W_out,   WoutT, 512, 256,  512);

        pack_tokens_kernel<<<(size_t)NTOK * D_MODEL / 256, 256, 0, stream>>>(x, U, dir);

        gemm_xz_kernel<<<dim3(NTOK / 128, 1024 / 64), 256, 0, stream>>>(U, WinT, XDT, Zbuf);

        conv_silu_kernel<<<(size_t)NTOK * D_INNER / 256, 256, 0, stream>>>(XDT, conv_w, conv_b, XCb);

        gemm_xproj_kernel<<<dim3(NTOK / 128, 3), 256, 0, stream>>>(XCb, WxT, DTIN, Bm, Cm);

        gemm_dt_kernel<<<dim3(NTOK / 128, 512 / 64), 256, 0, stream>>>(DTIN, WdtT, b_dt, XDT /* DT */);

        scan_kernel<<<(NSEQ * D_INNER) / 256, 256, 0, stream>>>(XDT, Bm, Cm, XCb, Zbuf, A_log, Dp, Yb);

        gemm_out_kernel<<<dim3(NTOK / 128, 256 / 64), 256, 0, stream>>>(Yb, WoutT, out, dir);
    }
}
